// MultiHeadAttention_79534204387890
// MI455X (gfx1250) — matrix operand order repaired and checked by reading, not compiled, not run on hardware
//
#include <hip/hip_runtime.h>
#include <hip/hip_bf16.h>

#define D_MODEL   1024
#define NUM_HEADS 16
#define HEAD_DIM  64
#define BATCH     2
#define SEQ       2048
#define NEG_INF   (-1.0e9f)

typedef __bf16 bf16_t;
typedef __attribute__((ext_vector_type(16))) bf16_t v16bf;
typedef __attribute__((ext_vector_type(8)))  bf16_t bf16x8;
typedef __attribute__((ext_vector_type(8)))  float  v8f;
typedef __attribute__((ext_vector_type(4)))  float  f32x4;

// ---- WMMA operand helpers (layouts per CDNA5 ISA 7.12.2, wave32) ----
// A (16x32 bf16): lane l<16 -> row l, K {0..7,16..23}; lane l>=16 -> row l-16, K {8..15,24..31}
// B (32x16 bf16): lane n<16 -> col n, K {0..7,16..23}; lane n>=16 -> col n-16, K {8..15,24..31}
// C (16x16 f32):  VGPR i -> (M=i, N=lane) lanes 0-15 ; (M=i+8, N=lane-16) lanes 16-31

__device__ __forceinline__ v16bf cat8(bf16x8 lo, bf16x8 hi) {
  return __builtin_shufflevector(lo, hi, 0,1,2,3,4,5,6,7,8,9,10,11,12,13,14,15);
}

__device__ __forceinline__ v16bf load_a_bf16(const bf16_t* rowp, int off8) {
  bf16x8 lo = *(const bf16x8*)(rowp + off8);
  bf16x8 hi = *(const bf16x8*)(rowp + 16 + off8);
  return cat8(lo, hi);
}

__device__ __forceinline__ v16bf load_a_f32_cvt(const float* rowp, int off8) {
  f32x4 c0 = *(const f32x4*)(rowp + off8 + 0);
  f32x4 c1 = *(const f32x4*)(rowp + off8 + 4);
  f32x4 c2 = *(const f32x4*)(rowp + off8 + 16);
  f32x4 c3 = *(const f32x4*)(rowp + off8 + 20);
  v16bf a;
#pragma unroll
  for (int j = 0; j < 4; ++j) {
    a[j]      = (bf16_t)c0[j];
    a[4 + j]  = (bf16_t)c1[j];
    a[8 + j]  = (bf16_t)c2[j];
    a[12 + j] = (bf16_t)c3[j];
  }
  return a;
}

// ---- Kernel 1: LDS-tiled transpose + convert: Wt[out][in] (bf16) = W[in][out] (f32) ----
// block (32,8), each block handles a 32x32 tile; both global accesses coalesced.
__global__ void wtrans_kernel(const float* __restrict__ W, bf16_t* __restrict__ Wt) {
  __shared__ float tile[32][33];              // +1 pad: no LDS bank conflicts
  const int tx = threadIdx.x;                 // 0..31
  const int ty = threadIdx.y;                 // 0..7
  const int i0 = blockIdx.x * 32;             // 'in' base
  const int o0 = blockIdx.y * 32;             // 'out' base
#pragma unroll
  for (int r = 0; r < 32; r += 8)             // read W[in][out], coalesced in 'out'
    tile[ty + r][tx] = W[(size_t)(i0 + ty + r) * D_MODEL + (o0 + tx)];
  __syncthreads();
#pragma unroll
  for (int r = 0; r < 32; r += 8)             // write Wt[out][in], coalesced in 'in'
    Wt[(size_t)(o0 + ty + r) * D_MODEL + (i0 + tx)] = (bf16_t)tile[tx][ty + r];
}

// ---- Kernel 2/5: GEMM  C[M=4096,N=1024] = A[M,1024] @ Wt^T + bias ----
// Each wave computes a 16x64 output tile (4 accumulators) -> 4x A reuse.
// out_mode: 0 = bf16 [B,H,S,Dh] (q/k), 1 = bf16 [B,H,Dh,S] (v transposed), 2 = f32 [B,S,D]
__global__ void proj_gemm_kernel(const void* __restrict__ Ain, int a_is_bf16,
                                 const bf16_t* __restrict__ Wt,
                                 const float* __restrict__ bias,
                                 void* __restrict__ Out, int out_mode) {
  const int lane   = threadIdx.x;                              // 0..31
  const int mt     = blockIdx.x;                               // 0..255
  const int ngrp   = blockIdx.y * blockDim.y + threadIdx.y;    // 0..15
  const int mbase  = mt * 16;
  const int nbase0 = ngrp * 64;
  const int l15    = lane & 15;
  const int colsel = lane >> 4;
  const int off8   = colsel * 8;
  const int rowA   = mbase + l15;

  v8f acc[4];
#pragma unroll
  for (int t = 0; t < 4; ++t)
#pragma unroll
    for (int j = 0; j < 8; ++j) acc[t][j] = 0.0f;

  const bf16_t* bp0 = Wt + (size_t)(nbase0 + l15) * D_MODEL + off8;

  if (a_is_bf16) {
    const bf16_t* ap = (const bf16_t*)Ain + (size_t)rowA * D_MODEL;
    for (int kb = 0; kb < D_MODEL; kb += 32) {
      if (kb + 32 < D_MODEL) __builtin_prefetch(ap + kb + 32, 0, 1);
      v16bf a = load_a_bf16(ap + kb, off8);
#pragma unroll
      for (int t = 0; t < 4; ++t) {
        v16bf b = load_a_bf16(bp0 + (size_t)t * 16 * D_MODEL + kb, 0);
        acc[t] = __builtin_amdgcn_wmma_f32_16x16x32_bf16(false, a, false, b,
                                                         (short)0, acc[t], false, false);
      }
    }
  } else {
    const float* ap = (const float*)Ain + (size_t)rowA * D_MODEL;
    for (int kb = 0; kb < D_MODEL; kb += 32) {
      if (kb + 32 < D_MODEL) __builtin_prefetch(ap + kb + 32, 0, 1);
      v16bf a = load_a_f32_cvt(ap + kb, off8);
#pragma unroll
      for (int t = 0; t < 4; ++t) {
        v16bf b = load_a_bf16(bp0 + (size_t)t * 16 * D_MODEL + kb, 0);
        acc[t] = __builtin_amdgcn_wmma_f32_16x16x32_bf16(false, a, false, b,
                                                         (short)0, acc[t], false, false);
      }
    }
  }

#pragma unroll
  for (int t = 0; t < 4; ++t) {
    const int   col = nbase0 + t * 16 + l15;
    const float bv  = bias[col];
#pragma unroll
    for (int i = 0; i < 8; ++i) {
      const int   row = mbase + i + 8 * colsel;   // 0..4095 (= b*SEQ + s)
      const float v   = acc[t][i] + bv;
      const int   bb  = row >> 11, s = row & (SEQ - 1);
      const int   h   = col >> 6,  dh = col & 63;
      if (out_mode == 0) {
        ((bf16_t*)Out)[(((size_t)bb * NUM_HEADS + h) * SEQ + s) * HEAD_DIM + dh] = (bf16_t)v;
      } else if (out_mode == 1) {
        ((bf16_t*)Out)[(((size_t)bb * NUM_HEADS + h) * HEAD_DIM + dh) * SEQ + s] = (bf16_t)v;
      } else {
        ((float*)Out)[(size_t)row * D_MODEL + col] = v;
      }
    }
  }
}

// ---- Kernel 3: scores = (q @ k^T) * scale, causal mask -> -1e9, fp32 to d_out ----
// grid (128 qtiles, 8, 32 bh), block (32,4): each wave does a 16x64 tile (8 WMMAs)
__global__ void attn_scores_kernel(const bf16_t* __restrict__ qb,
                                   const bf16_t* __restrict__ kb,
                                   float* __restrict__ sc) {
  const int lane   = threadIdx.x;
  const int qt     = blockIdx.x;                               // 0..127
  const int ng     = blockIdx.y * blockDim.y + threadIdx.y;    // 0..31
  const int bh     = blockIdx.z;                               // 0..31
  const int qbase  = qt * 16;
  const int nbase0 = ng * 64;
  const int l15    = lane & 15, colsel = lane >> 4, off8 = colsel * 8;

  const bf16_t* qp = qb + (size_t)bh * SEQ * HEAD_DIM + (size_t)(qbase + l15) * HEAD_DIM;
  const bf16_t* kp = kb + (size_t)bh * SEQ * HEAD_DIM;
  float*        sp = sc + (size_t)bh * SEQ * SEQ;

  v8f acc[4];
#pragma unroll
  for (int t = 0; t < 4; ++t)
#pragma unroll
    for (int j = 0; j < 8; ++j) acc[t][j] = 0.0f;

  if (nbase0 <= qbase + 15) {   // tile group intersects causal region
#pragma unroll
    for (int kk = 0; kk < HEAD_DIM; kk += 32) {
      v16bf a = load_a_bf16(qp + kk, off8);
#pragma unroll
      for (int t = 0; t < 4; ++t) {
        const bf16_t* bp = kp + (size_t)(nbase0 + t * 16 + l15) * HEAD_DIM + kk;
        v16bf b = load_a_bf16(bp, off8);
        acc[t] = __builtin_amdgcn_wmma_f32_16x16x32_bf16(false, a, false, b,
                                                         (short)0, acc[t], false, false);
      }
    }
  }

  const float scale = 0.125f;   // 1/sqrt(64)
#pragma unroll
  for (int t = 0; t < 4; ++t) {
#pragma unroll
    for (int i = 0; i < 8; ++i) {
      const int row = qbase + i + 8 * colsel;
      const int cl  = nbase0 + t * 16 + l15;
      const float v = (cl <= row) ? acc[t][i] * scale : NEG_INF;
      sp[(size_t)row * SEQ + cl] = v;
    }
  }
}

// ---- Kernel 4: rowwise softmax over scores + P @ V -> combined (bf16 [B,S,D]) ----
// one wave per (bh, 16-row q block); grid 4096, block 32
__global__ void attn_softmax_pv_kernel(const float* __restrict__ sc,
                                       const bf16_t* __restrict__ vT,
                                       bf16_t* __restrict__ comb) {
  const int lane  = threadIdx.x;
  const int blk   = blockIdx.x;
  const int bh    = blk >> 7;
  const int qt    = blk & 127;
  const int qbase = qt * 16;
  const int kend  = (qbase + 16 + 31) & ~31;   // causal limit rounded to k-tile

  const float* sbase = sc + (size_t)bh * SEQ * SEQ + (size_t)qbase * SEQ;

  __shared__ float sm[16];
  __shared__ float sinv[16];

  // Phase 1: online max + sum-of-exp, two lanes per row
  {
    const int    row  = lane >> 1;
    const int    half = lane & 1;
    const float* srow = sbase + (size_t)row * SEQ;
    float m = -3.0e38f, l = 0.0f;
    for (int j = half; 4 * j < kend; j += 2) {
      f32x4 v  = *(const f32x4*)(srow + 4 * j);
      float cm = fmaxf(fmaxf(v[0], v[1]), fmaxf(v[2], v[3]));
      if (cm > m) { l *= __expf(m - cm); m = cm; }
      l += __expf(v[0] - m) + __expf(v[1] - m) + __expf(v[2] - m) + __expf(v[3] - m);
    }
    float mo = __shfl_xor(m, 1, 32);
    float lo = __shfl_xor(l, 1, 32);
    float mm = fmaxf(m, mo);
    float ll = l * __expf(m - mm) + lo * __expf(mo - mm);
    if (half == 0) { sm[row] = mm; sinv[row] = 1.0f / ll; }
  }
  __syncthreads();

  // Phase 2: attn = softmax(scores) @ V, streamed over 32-wide k tiles
  const int     l15    = lane & 15, colsel = lane >> 4, off8 = colsel * 8;
  const float   mrow   = sm[l15];
  const bf16_t* vp     = vT + (size_t)bh * HEAD_DIM * SEQ;
  const float*  sp     = sbase + (size_t)l15 * SEQ;

  v8f acc[4];
#pragma unroll
  for (int t = 0; t < 4; ++t)
#pragma unroll
    for (int j = 0; j < 8; ++j) acc[t][j] = 0.0f;

  for (int kbk = 0; kbk < kend; kbk += 32) {
    f32x4 c0 = *(const f32x4*)(sp + kbk + off8 + 0);
    f32x4 c1 = *(const f32x4*)(sp + kbk + off8 + 4);
    f32x4 c2 = *(const f32x4*)(sp + kbk + off8 + 16);
    f32x4 c3 = *(const f32x4*)(sp + kbk + off8 + 20);
    v16bf a;
#pragma unroll
    for (int j = 0; j < 4; ++j) {
      a[j]      = (bf16_t)__expf(c0[j] - mrow);
      a[4 + j]  = (bf16_t)__expf(c1[j] - mrow);
      a[8 + j]  = (bf16_t)__expf(c2[j] - mrow);
      a[12 + j] = (bf16_t)__expf(c3[j] - mrow);
    }
#pragma unroll
    for (int t = 0; t < 4; ++t) {
      const bf16_t* bp = vp + (size_t)(t * 16 + l15) * SEQ + kbk;
      v16bf b = load_a_bf16(bp, off8);
      acc[t] = __builtin_amdgcn_wmma_f32_16x16x32_bf16(false, a, false, b,
                                                       (short)0, acc[t], false, false);
    }
  }

  const int b = bh >> 4, h = bh & 15;
#pragma unroll
  for (int t = 0; t < 4; ++t) {
#pragma unroll
    for (int i = 0; i < 8; ++i) {
      const int   r   = i + 8 * colsel;
      const float inv = sinv[r];
      const int   s   = qbase + r;
      const int   col = h * HEAD_DIM + t * 16 + l15;
      comb[((size_t)b * SEQ + s) * D_MODEL + col] = (bf16_t)(acc[t][i] * inv);
    }
  }
}

extern "C" void kernel_launch(void* const* d_in, const int* in_sizes, int n_in,
                              void* d_out, int out_size, void* d_ws, size_t ws_size,
                              hipStream_t stream) {
  (void)in_sizes; (void)n_in; (void)out_size; (void)ws_size;

  const float* Q   = (const float*)d_in[0];
  const float* Kin = (const float*)d_in[1];
  const float* Vin = (const float*)d_in[2];
  // d_in[3] = mask (tril) -> causal mask computed from indices
  const float* w_q = (const float*)d_in[4];  const float* b_q = (const float*)d_in[5];
  const float* w_k = (const float*)d_in[6];  const float* b_k = (const float*)d_in[7];
  const float* w_v = (const float*)d_in[8];  const float* b_v = (const float*)d_in[9];
  const float* w_o = (const float*)d_in[10]; const float* b_o = (const float*)d_in[11];

  float* out    = (float*)d_out;                               // [B,S,D]
  float* scores = out + (size_t)BATCH * SEQ * D_MODEL;         // [B,H,S,S]

  const size_t MB = (size_t)1 << 20;
  char* ws = (char*)d_ws;
  bf16_t* WqT  = (bf16_t*)(ws + 0 * MB);   // [out,in] 2 MB each
  bf16_t* WkT  = (bf16_t*)(ws + 2 * MB);
  bf16_t* WvT  = (bf16_t*)(ws + 4 * MB);
  bf16_t* WoT  = (bf16_t*)(ws + 6 * MB);
  bf16_t* qbuf = (bf16_t*)(ws + 8 * MB);   // [B,H,S,Dh]  8 MB
  bf16_t* kbuf = (bf16_t*)(ws + 16 * MB);  // [B,H,S,Dh]  8 MB
  bf16_t* vTb  = (bf16_t*)(ws + 24 * MB);  // [B,H,Dh,S]  8 MB
  bf16_t* comb = (bf16_t*)(ws + 32 * MB);  // [B,S,D]     8 MB

  // 1. weight transpose + bf16 convert (LDS-tiled, coalesced both ways)
  {
    dim3 tg(32, 32);
    dim3 tb(32, 8);
    wtrans_kernel<<<tg, tb, 0, stream>>>(w_q, WqT);
    wtrans_kernel<<<tg, tb, 0, stream>>>(w_k, WkT);
    wtrans_kernel<<<tg, tb, 0, stream>>>(w_v, WvT);
    wtrans_kernel<<<tg, tb, 0, stream>>>(w_o, WoT);
  }

  // 2. input projections (fp32 A, converted inline); 16x64 tile per wave
  dim3 gblk(32, 4);
  dim3 ggrid(256, 4);
  proj_gemm_kernel<<<ggrid, gblk, 0, stream>>>((const void*)Q,   0, WqT, b_q, (void*)qbuf, 0);
  proj_gemm_kernel<<<ggrid, gblk, 0, stream>>>((const void*)Kin, 0, WkT, b_k, (void*)kbuf, 0);
  proj_gemm_kernel<<<ggrid, gblk, 0, stream>>>((const void*)Vin, 0, WvT, b_v, (void*)vTb,  1);

  // 3. masked scores -> d_out (second output)
  attn_scores_kernel<<<dim3(128, 8, 32), dim3(32, 4), 0, stream>>>(qbuf, kbuf, scores);

  // 4. softmax + P@V -> combined
  attn_softmax_pv_kernel<<<dim3(4096), dim3(32), 0, stream>>>(scores, vTb, comb);

  // 5. output projection (bf16 A) -> d_out (first output)
  proj_gemm_kernel<<<ggrid, gblk, 0, stream>>>((const void*)comb, 1, WoT, b_o, (void*)out, 2);
}
